// GeoAttn_90941637525615
// MI455X (gfx1250) — matrix operand order repaired and checked by reading, not compiled, not run on hardware
//
#include <hip/hip_runtime.h>
#include <math.h>

// ---------------------------------------------------------------------------
// Types for CDNA5 WMMA (wave32): D(16x16 f32) = A(16x32 bf16) x B(32x16 bf16) + C
// ---------------------------------------------------------------------------
typedef __attribute__((ext_vector_type(16))) __bf16 v16bf;
typedef __attribute__((ext_vector_type(8)))  __bf16 v8bf;
typedef __attribute__((ext_vector_type(8)))  float  v8f;

#define LL    256   // sequence length
#define NV    128   // h_V feature dim
#define NE    128   // h_E feature dim
#define NH    128   // hidden dim
#define ITILE 8     // queries per workgroup
#define PROWS 128   // pair rows per tile = ITILE * 16 keys

// ---- weight-pack tile table (each tile = 32x16 bf16 = 512 halfs, B-fragment order)
#define T_B1  0     // 384x128 -> 12 ktiles x 8 ntiles = 96
#define T_B2  96    // 128x128 -> 32
#define T_B3  128   // 128x4 (N padded to 16) -> 4
#define T_WV  132   // 128x128 -> 32
#define T_WO  164   // 128x128 -> 32
#define T_GT  196   // 128x128 -> 32
#define T_TOT 228

// ---- dynamic LDS layout (bytes); sXf aliases sVj (dead after sweep 1)
#define OFF_VI  0                         // 8x128 bf16   (query hV tile)     2048
#define OFF_VJ  2048                      // 16x128 bf16  (key hV tile)       4096
#define OFF_XF  2048                      // 16x128 bf16  (alias, final hV)   4096
#define OFF_E   6144                      // 128x128 bf16 (h_E pair tile)    32768
#define OFF_X   38912                     // 128x128 bf16 (layer1 out)       32768
#define OFF_Y   71680                     // 128x128 bf16 (layer2 out)       32768
#define OFF_LA  104448                    // 8x256x4 f32  (logits/attend)    32768
#define OFF_ACC 137216                    // 8x128 f32    (hV accumulator)    4096
#define OFF_CI  141312                    // 16x128 f32   (hV_i @ B1 chunk)   8192
#define OFF_CJ  149504                    // 16x128 f32   (hV_j @ B1 chunk)   8192
#define SMEM_BYTES 157696                 // 154 KB -> 2 workgroups / 320KB WGP

static __device__ __forceinline__ __bf16 f2bf(float f) {
  union { float f; unsigned u; } x; x.f = f;
  unsigned r = x.u + 0x7FFFu + ((x.u >> 16) & 1u);       // round-to-nearest-even
  unsigned short s = (unsigned short)(r >> 16);
  union { unsigned short s; __bf16 b; } y; y.s = s;
  return y.b;
}

// A-fragment (16x32 bf16): lane m = lane&15; lanes<16 hold K {0..7,16..23},
// lanes>=16 hold K {8..15,24..31}. 'row' points at A row m, col 0 in LDS.
static __device__ __forceinline__ v16bf load_a_frag(const __bf16* row, int k0, int hi) {
  v8bf lo = *reinterpret_cast<const v8bf*>(row + k0 + hi * 8);
  v8bf hb = *reinterpret_cast<const v8bf*>(row + k0 + hi * 8 + 16);
  v16bf a;
#pragma unroll
  for (int i = 0; i < 8; ++i) { a[i] = lo[i]; a[i + 8] = hb[i]; }
  return a;
}

// B-fragment: pre-swizzled in global scratch, 16 halfs (32B) contiguous per lane.
static __device__ __forceinline__ v16bf load_b_frag(const __bf16* tile, int lane) {
  return *reinterpret_cast<const v16bf*>(tile + (size_t)lane * 16);
}

static __device__ __forceinline__ v8f wmma_bf16(v16bf a, v16bf b, v8f c) {
  return __builtin_amdgcn_wmma_f32_16x16x32_bf16(false, a, false, b, (short)0, c,
                                                 false, false);
}

// ---------------------------------------------------------------------------
// Pack fp32 row-major weights [K][N] into bf16 WMMA-B fragment order.
// One block per 32x16 tile, 512 threads = one element each.
// Lane<16: k = kt*32+{0..7,16..23}[slot],  n = nt*16+lane
// Lane>=16: k = kt*32+{8..15,24..31}[slot], n = nt*16+lane-16
// ---------------------------------------------------------------------------
__global__ __launch_bounds__(512) void pack_weights(
    const float* __restrict__ B1, const float* __restrict__ B2,
    const float* __restrict__ B3, const float* __restrict__ Wv,
    const float* __restrict__ Wo, const float* __restrict__ Gt,
    __bf16* __restrict__ dst) {
  int t = blockIdx.x;
  int e = threadIdx.x;
  const float* W; int K, Nlog, NT, local;
  if      (t < T_B2) { W = B1; K = 384; Nlog = 128; NT = 8; local = t - T_B1; }
  else if (t < T_B3) { W = B2; K = 128; Nlog = 128; NT = 8; local = t - T_B2; }
  else if (t < T_WV) { W = B3; K = 128; Nlog = 4;   NT = 1; local = t - T_B3; }
  else if (t < T_WO) { W = Wv; K = 128; Nlog = 128; NT = 8; local = t - T_WV; }
  else if (t < T_GT) { W = Wo; K = 128; Nlog = 128; NT = 8; local = t - T_WO; }
  else               { W = Gt; K = 128; Nlog = 128; NT = 8; local = t - T_GT; }
  int kt = local / NT, nt = local % NT;
  int lane = e >> 4, slot = e & 15;
  int k = kt * 32 + (slot >> 3) * 16 + (lane >= 16 ? 8 : 0) + (slot & 7);
  int n = nt * 16 + (lane & 15);
  float v = (n < Nlog && k < K) ? W[k * Nlog + n] : 0.0f;
  dst[(size_t)t * 512 + e] = f2bf(v);
}

// ---------------------------------------------------------------------------
// Fused geometric-attention kernel.
// Grid: B * (L/ITILE) = 128 blocks; 256 threads = 8 wave32; wave w owns M-tile w
// (pair rows p in [16w,16w+16), query ii = w, key jj = p&15).
// ---------------------------------------------------------------------------
__global__ __launch_bounds__(256) void geoattn_main(
    const float* __restrict__ hVp, const float* __restrict__ hE,
    const unsigned char* __restrict__ amask,
    const float* __restrict__ pWvb, const float* __restrict__ pB1b,
    const float* __restrict__ pB2b, const float* __restrict__ pB3b,
    const float* __restrict__ pGb,
    const __bf16* __restrict__ wp, float* __restrict__ out) {
  extern __shared__ __align__(16) char smem[];
  __bf16* sVi = (__bf16*)(smem + OFF_VI);
  __bf16* sVj = (__bf16*)(smem + OFF_VJ);
  __bf16* sXf = (__bf16*)(smem + OFF_XF);  // aliases sVj (dead by then)
  __bf16* sE  = (__bf16*)(smem + OFF_E);
  __bf16* sX  = (__bf16*)(smem + OFF_X);
  __bf16* sY  = (__bf16*)(smem + OFF_Y);
  float*  sLA = (float*)(smem + OFF_LA);   // [ii][j][head]
  float*  acc = (float*)(smem + OFF_ACC);  // [ii][col]
  float*  sCI = (float*)(smem + OFF_CI);   // [ii][col] = hV_i @ B1[0:128]
  float*  sCJ = (float*)(smem + OFF_CJ);   // [jj][col] = hV_j @ B1[256:384]

  const int tid  = threadIdx.x;
  const int lane = tid & 31;
  const int wave = tid >> 5;
  const int hi   = lane >> 4;
  const int ln   = lane & 15;
  const int b    = blockIdx.x >> 5;
  const int i0   = (blockIdx.x & 31) * ITILE;

  // ---- load query hV tile (8x128) to bf16, zero accumulator
  {
    int row = tid >> 5, c0 = (tid & 31) * 4;
    float4 v = *reinterpret_cast<const float4*>(hVp + ((size_t)b * LL + i0 + row) * NV + c0);
    sVi[row * NV + c0 + 0] = f2bf(v.x); sVi[row * NV + c0 + 1] = f2bf(v.y);
    sVi[row * NV + c0 + 2] = f2bf(v.z); sVi[row * NV + c0 + 3] = f2bf(v.w);
#pragma unroll
    for (int k = 0; k < 4; ++k) acc[tid + k * 256] = 0.0f;
  }
  __syncthreads();

  // ---- hoisted: contribI[ii][col] = hV_i @ B1[0:128]  (once per workgroup)
  //      wave w computes N-tile w; fragment rows m>=8 are garbage (never read).
  {
    const int nt = wave;
    const __bf16* aI = sVi + ln * NV;
    v8f c;
#pragma unroll
    for (int r = 0; r < 8; ++r) c[r] = 0.0f;
#pragma unroll
    for (int ks = 0; ks < 4; ++ks)
      c = wmma_bf16(load_a_frag(aI, ks * 32, hi),
                    load_b_frag(wp + (size_t)(T_B1 + ks * 8 + nt) * 512, lane), c);
#pragma unroll
    for (int r = 0; r < 8; ++r)
      sCI[(r + hi * 8) * NH + nt * 16 + ln] = c[r];
  }

  // =======================================================================
  // Sweep 1: masked per-head logits for all 256 keys
  // =======================================================================
  for (int j0 = 0; j0 < LL; j0 += 16) {
    __syncthreads();  // previous tile's sE/sVj/sCJ fully consumed
    {  // h_E tile: 128 pair rows x 128 feats, 2 threads per row
      int p = tid >> 1, half = tid & 1, c0 = half * 64;
      int ii = p >> 4, jj = p & 15;
      const float4* src = reinterpret_cast<const float4*>(
          hE + (((size_t)b * LL + (i0 + ii)) * LL + (j0 + jj)) * NE + c0);
#pragma unroll
      for (int q = 0; q < 16; ++q) {
        float4 v = src[q];
        int c = p * NE + c0 + q * 4;
        sE[c + 0] = f2bf(v.x); sE[c + 1] = f2bf(v.y);
        sE[c + 2] = f2bf(v.z); sE[c + 3] = f2bf(v.w);
      }
    }
    if (tid < 128) {  // key hV tile 16x128
      int row = tid >> 3, c0 = (tid & 7) * 16;
      const float4* src = reinterpret_cast<const float4*>(
          hVp + ((size_t)b * LL + j0 + row) * NV + c0);
#pragma unroll
      for (int q = 0; q < 4; ++q) {
        float4 v = src[q];
        int c = row * NV + c0 + q * 4;
        sVj[c + 0] = f2bf(v.x); sVj[c + 1] = f2bf(v.y);
        sVj[c + 2] = f2bf(v.z); sVj[c + 3] = f2bf(v.w);
      }
    }
    __syncthreads();

    // ---- hoisted: contribJ[jj][col] = hV_j @ B1[256:384] (once per j-tile,
    //      shared by all 8 waves; wave w computes N-tile w)
    {
      const int nt = wave;
      const __bf16* aJ = sVj + ln * NV;
      v8f c;
#pragma unroll
      for (int r = 0; r < 8; ++r) c[r] = 0.0f;
#pragma unroll
      for (int ks = 0; ks < 4; ++ks)
        c = wmma_bf16(load_a_frag(aJ, ks * 32, hi),
                      load_b_frag(wp + (size_t)(T_B1 + (ks + 8) * 8 + nt) * 512, lane), c);
#pragma unroll
      for (int r = 0; r < 8; ++r)
        sCJ[(r + hi * 8) * NH + nt * 16 + ln] = c[r];
    }
    __syncthreads();  // contribJ visible to all waves

    const int mt = wave;  // M-tile == query ii
    // ---- GEMM1: relu(b1 + contribI + contribJ + hE @ B1[128:256])
    {
      const __bf16* aE = sE + (mt * 16 + ln) * NE;      // pair row p
      for (int nt = 0; nt < 8; ++nt) {
        const int col = nt * 16 + ln;
        const float base = pB1b[col] + sCI[mt * NH + col];
        v8f c;
#pragma unroll
        for (int r = 0; r < 8; ++r) c[r] = base + sCJ[(r + hi * 8) * NH + col];
#pragma unroll
        for (int ks = 0; ks < 4; ++ks)
          c = wmma_bf16(load_a_frag(aE, ks * 32, hi),
                        load_b_frag(wp + (size_t)(T_B1 + (ks + 4) * 8 + nt) * 512, lane), c);
#pragma unroll
        for (int r = 0; r < 8; ++r) {
          int row = mt * 16 + r + hi * 8;
          sX[row * NH + col] = f2bf(fmaxf(c[r], 0.0f));
        }
      }
    }
    // ---- GEMM2: relu(X1 @ B2 + b2)  (own M-tile rows only -> no barrier)
    {
      const __bf16* aX = sX + (mt * 16 + ln) * NH;
      for (int nt = 0; nt < 8; ++nt) {
        v8f c; float bv = pB2b[nt * 16 + ln];
#pragma unroll
        for (int r = 0; r < 8; ++r) c[r] = bv;
#pragma unroll
        for (int ks = 0; ks < 4; ++ks)
          c = wmma_bf16(load_a_frag(aX, ks * 32, hi),
                        load_b_frag(wp + (size_t)(T_B2 + ks * 8 + nt) * 512, lane), c);
#pragma unroll
        for (int r = 0; r < 8; ++r) {
          int row = mt * 16 + r + hi * 8;
          sY[row * NH + nt * 16 + ln] = f2bf(fmaxf(c[r], 0.0f));
        }
      }
    }
    // ---- GEMM3: logits = X2 @ B3 + b3 (N padded 4->16), scale + mask
    {
      const __bf16* aY = sY + (mt * 16 + ln) * NH;
      v8f c; float bv = (ln < 4) ? pB3b[ln] : 0.0f;
#pragma unroll
      for (int r = 0; r < 8; ++r) c[r] = bv;
#pragma unroll
      for (int ks = 0; ks < 4; ++ks)
        c = wmma_bf16(load_a_frag(aY, ks * 32, hi),
                      load_b_frag(wp + (size_t)(T_B3 + ks) * 512, lane), c);
      if (ln < 4) {
#pragma unroll
        for (int r = 0; r < 8; ++r) {
          int jabs = j0 + r + hi * 8;
          float lg = c[r] * 0.17677669529663687f;  // 1/sqrt(32)
          float mk = (float)amask[((size_t)b * LL + (i0 + mt)) * LL + jabs];
          lg -= (1.0f - mk) * 999999999.0f;
          sLA[((size_t)mt * LL + jabs) * 4 + ln] = lg;
        }
      }
    }
  }
  __syncthreads();

  // ---- softmax over keys j for each (query ii, head h)
  if (tid < 32) {
    int ii = tid >> 2, h = tid & 3;
    float* row = sLA + (size_t)ii * LL * 4 + h;
    float mx = -1e30f;
    for (int j = 0; j < LL; ++j) mx = fmaxf(mx, row[j * 4]);
    float s = 0.0f;
    for (int j = 0; j < LL; ++j) { float e = __expf(row[j * 4] - mx); row[j * 4] = e; s += e; }
    float inv = 1.0f / fmaxf(s, 1e-20f);
    for (int j = 0; j < LL; ++j) row[j * 4] *= inv;
  }
  __syncthreads();

  // =======================================================================
  // Sweep 2: V = gelu(h_E @ Wv + bv); acc += attend * V
  // =======================================================================
  for (int j0 = 0; j0 < LL; j0 += 16) {
    __syncthreads();
    {
      int p = tid >> 1, half = tid & 1, c0 = half * 64;
      int ii = p >> 4, jj = p & 15;
      const float4* src = reinterpret_cast<const float4*>(
          hE + (((size_t)b * LL + (i0 + ii)) * LL + (j0 + jj)) * NE + c0);
#pragma unroll
      for (int q = 0; q < 16; ++q) {
        float4 v = src[q];
        int c = p * NE + c0 + q * 4;
        sE[c + 0] = f2bf(v.x); sE[c + 1] = f2bf(v.y);
        sE[c + 2] = f2bf(v.z); sE[c + 3] = f2bf(v.w);
      }
    }
    __syncthreads();

    const int mt = wave;
    const __bf16* aE = sE + (mt * 16 + ln) * NE;
    for (int nt = 0; nt < 8; ++nt) {
      v8f c; float bv = pWvb[nt * 16 + ln];
#pragma unroll
      for (int r = 0; r < 8; ++r) c[r] = bv;
#pragma unroll
      for (int ks = 0; ks < 4; ++ks)
        c = wmma_bf16(load_a_frag(aE, ks * 32, hi),
                      load_b_frag(wp + (size_t)(T_WV + ks * 8 + nt) * 512, lane), c);
      const int head = nt >> 1;  // col/32
      float psum = 0.0f;
#pragma unroll
      for (int r = 0; r < 8; ++r) {
        float x = c[r];
        float g = 0.5f * x * (1.0f + erff(x * 0.70710678118654752f));  // exact gelu
        int jj = r + hi * 8;
        psum += sLA[((size_t)mt * LL + j0 + jj) * 4 + head] * g;
      }
      atomicAdd(&acc[mt * NH + nt * 16 + ln], psum);  // ds_add_f32, lanes n & n+16
    }
  }
  __syncthreads();

  // ---- acc (8x128 f32) -> sXf (16x128 bf16, zero-padded rows 8..15)
#pragma unroll
  for (int k = 0; k < 8; ++k) {
    int idx = tid + k * 256;
    sXf[idx] = f2bf(idx < ITILE * NH ? acc[idx] : 0.0f);
  }
  __syncthreads();

  // ---- final: dh = (hV_acc @ Wo) * sigmoid(hV_acc @ gate + gb); out = h_V + dh
  {
    const int nt = wave;  // 8 waves x 8 N-tiles
    const __bf16* aH = sXf + ln * NH;
    v8f cg, co; float gb = pGb[nt * 16 + ln];
#pragma unroll
    for (int r = 0; r < 8; ++r) { cg[r] = gb; co[r] = 0.0f; }
#pragma unroll
    for (int ks = 0; ks < 4; ++ks) {
      v16bf a = load_a_frag(aH, ks * 32, hi);
      cg = wmma_bf16(a, load_b_frag(wp + (size_t)(T_GT + ks * 8 + nt) * 512, lane), cg);
      co = wmma_bf16(a, load_b_frag(wp + (size_t)(T_WO + ks * 8 + nt) * 512, lane), co);
    }
#pragma unroll
    for (int r = 0; r < 8; ++r) {
      int ii = r + hi * 8;
      if (ii < ITILE) {
        int col = nt * 16 + ln;
        float g = 1.0f / (1.0f + __expf(-cg[r]));
        size_t idx = ((size_t)b * LL + i0 + ii) * NV + col;
        out[idx] = hVp[idx] + co[r] * g;
      }
    }
  }
}

// ---------------------------------------------------------------------------
extern "C" void kernel_launch(void* const* d_in, const int* in_sizes, int n_in,
                              void* d_out, int out_size, void* d_ws, size_t ws_size,
                              hipStream_t stream) {
  (void)in_sizes; (void)n_in; (void)out_size; (void)ws_size;
  const float* hV  = (const float*)d_in[0];
  const float* hE  = (const float*)d_in[1];
  const unsigned char* am = (const unsigned char*)d_in[2];
  const float* Wvw = (const float*)d_in[3];
  const float* Wvb = (const float*)d_in[4];
  const float* B1w = (const float*)d_in[5];
  const float* B1b = (const float*)d_in[6];
  const float* B2w = (const float*)d_in[7];
  const float* B2b = (const float*)d_in[8];
  const float* B3w = (const float*)d_in[9];
  const float* B3b = (const float*)d_in[10];
  const float* Wow = (const float*)d_in[11];
  const float* Gtw = (const float*)d_in[12];
  const float* Gtb = (const float*)d_in[13];
  __bf16* wp = (__bf16*)d_ws;  // 228 tiles * 1 KB = 233 KB of scratch

  pack_weights<<<dim3(T_TOT), dim3(512), 0, stream>>>(B1w, B2w, B3w, Wvw, Wow, Gtw, wp);

  hipFuncSetAttribute(reinterpret_cast<const void*>(geoattn_main),
                      hipFuncAttributeMaxDynamicSharedMemorySize, SMEM_BYTES);
  geoattn_main<<<dim3(4 * (LL / ITILE)), dim3(256), SMEM_BYTES, stream>>>(
      hV, hE, am, Wvb, B1b, B2b, B3b, Gtb, wp, (float*)d_out);
}
